// FieldWeightedFactorizationMachineModel_24996709663382
// MI455X (gfx1250) — compile-verified, matrix operand order checked by reading, not run on hardware
//
#include <hip/hip_runtime.h>

#define F_FIELDS 39
#define D_DIM    64

typedef __bf16 v16bf __attribute__((ext_vector_type(16)));
typedef __bf16 v2bf  __attribute__((ext_vector_type(2)));
typedef float  v8f   __attribute__((ext_vector_type(8)));

union FragU { v16bf v; v2bf p[8]; };

__device__ __forceinline__ v2bf pk_bf16(float x, float y) {
#if __has_builtin(__builtin_amdgcn_cvt_pk_bf16_f32)
    return __builtin_amdgcn_cvt_pk_bf16_f32(x, y);   // v_cvt_pk_bf16_f32 (RNE)
#else
    v2bf r; r.x = (__bf16)x; r.y = (__bf16)y; return r;
#endif
}

__global__ __launch_bounds__(256) void fwfm_wmma_kernel(
    const int*   __restrict__ x,      // (B, 39)
    const float* __restrict__ emb,    // (V, 64)
    const float* __restrict__ bias,   // (V, 1)
    const float* __restrict__ W,      // (39, 39)
    const float* __restrict__ w0,     // (1,)
    float*       __restrict__ out,    // (B,)
    int Brows)
{
    const int lane   = threadIdx.x & 31;
    const int wave   = (int)((blockIdx.x * blockDim.x + threadIdx.x) >> 5);
    const int nwaves = (int)((gridDim.x * blockDim.x) >> 5);

    // Upper-triangular tile pairs of the 3x3 (48x48-padded) Gram tile grid.
    const int TI[6] = {0,0,0,1,1,2};
    const int TJ[6] = {0,1,2,1,2,2};

    // ---- Loop-invariant: masked interaction weights, pre-gathered into the
    //      16x16 f32 C/D layout: lanes 0-15 -> N=lane, lanes 16-31 -> N=lane-16;
    //      VGPR r -> M=r (low half) or M=8+r (high half).
    const int n_c  = lane & 15;
    const int m_hi = (lane >> 4) * 8;
    float wreg[6][8];
    #pragma unroll
    for (int tp = 0; tp < 6; ++tp) {
        #pragma unroll
        for (int r = 0; r < 8; ++r) {
            int i = 16 * TI[tp] + m_hi + r;
            int j = 16 * TJ[tp] + n_c;
            // keep only i < j < F (strict upper triangle, in-bounds)
            wreg[tp][r] = (i < j && j < F_FIELDS) ? W[i * F_FIELDS + j] : 0.0f;
        }
    }

    // bf16 16x32 A-operand layout: lane m (low half) holds K = {0..7, 16..23},
    // lane m+16 holds K = {8..15, 24..31}; second K-half adds 32.
    const int rrow   = lane & 15;
    const int kbase0 = (lane & 16) ? 8 : 0;

    for (int b = wave; b < Brows; b += nwaves) {
        const int* xb = x + (size_t)b * F_FIELDS;

        // ---- Gather embedding fragments frag[tile][khalf] (bf16).
        FragU frag[3][2];
        #pragma unroll
        for (int t = 0; t < 3; ++t) {
            const int  row   = 16 * t + rrow;
            const bool valid = (row < F_FIELDS);
            const int  idx   = valid ? xb[row] : 0;
            const float* erow = emb + (size_t)idx * D_DIM;
            #pragma unroll
            for (int h = 0; h < 2; ++h) {
                const int kb = kbase0 + 32 * h;
                FragU& f = frag[t][h];
                if (valid) {
                    const float4* p0 = (const float4*)(erow + kb);        // K = kb..kb+7
                    const float4* p1 = (const float4*)(erow + kb + 16);   // K = kb+16..kb+23
                    float4 a0 = p0[0], a1 = p0[1];
                    float4 b0 = p1[0], b1 = p1[1];
                    f.p[0] = pk_bf16(a0.x, a0.y);
                    f.p[1] = pk_bf16(a0.z, a0.w);
                    f.p[2] = pk_bf16(a1.x, a1.y);
                    f.p[3] = pk_bf16(a1.z, a1.w);
                    f.p[4] = pk_bf16(b0.x, b0.y);
                    f.p[5] = pk_bf16(b0.z, b0.w);
                    f.p[6] = pk_bf16(b1.x, b1.y);
                    f.p[7] = pk_bf16(b1.z, b1.w);
                } else {
                    v2bf z = pk_bf16(0.0f, 0.0f);
                    #pragma unroll
                    for (int s = 0; s < 8; ++s) f.p[s] = z;
                }
            }
        }

        // ---- Gram tiles via bf16 WMMA (same fragment as A and B -> E * E^T),
        //      contracted on the fly with the masked W in C-layout.
        float wsum = 0.0f;
        #pragma unroll
        for (int tp = 0; tp < 6; ++tp) {
            v8f acc = {};
            acc = __builtin_amdgcn_wmma_f32_16x16x32_bf16(
                false, frag[TI[tp]][0].v, false, frag[TJ[tp]][0].v,
                (short)0, acc, false, false);
            acc = __builtin_amdgcn_wmma_f32_16x16x32_bf16(
                false, frag[TI[tp]][1].v, false, frag[TJ[tp]][1].v,
                (short)0, acc, false, false);
            #pragma unroll
            for (int r = 0; r < 8; ++r) wsum += wreg[tp][r] * acc[r];
        }

        // ---- Bias gathers: lane covers field=lane (always <39) and field=lane+32.
        wsum += bias[(size_t)xb[lane]];
        if (lane + 32 < F_FIELDS) wsum += bias[(size_t)xb[lane + 32]];

        // ---- wave32 reduction
        #pragma unroll
        for (int off = 16; off >= 1; off >>= 1)
            wsum += __shfl_xor(wsum, off, 32);

        if (lane == 0) out[b] = w0[0] + wsum;
    }
}

extern "C" void kernel_launch(void* const* d_in, const int* in_sizes, int n_in,
                              void* d_out, int out_size, void* d_ws, size_t ws_size,
                              hipStream_t stream) {
    const int*   x    = (const int*)  d_in[0];
    const float* emb  = (const float*)d_in[1];
    const float* bias = (const float*)d_in[2];
    const float* W    = (const float*)d_in[3];
    const float* w0   = (const float*)d_in[4];
    float*       out  = (float*)d_out;

    const int Brows = in_sizes[0] / F_FIELDS;      // 32768
    const int block = 256;                         // 8 wave32 per workgroup
    int grid = (Brows + 7) / 8;                    // one row per wave
    if (grid < 1) grid = 1;

    fwfm_wmma_kernel<<<grid, block, 0, stream>>>(x, emb, bias, W, w0, out, Brows);
}
